// ERPBlock_62414464745682
// MI455X (gfx1250) — matrix operand order repaired and checked by reading, not compiled, not run on hardware
//
#include <hip/hip_runtime.h>
#include <math.h>

// ---------------------------------------------------------------------------
// ERP block for MI455X (gfx1250, wave32).
// Heavy 3x3 convs (base conv + deformable einsum) run as implicit GEMM on
// V_WMMA_F32_16X16X32_BF16 with A = pre-packed weights, B = activations read
// from an NHWC bf16 copy (32B contiguous per lane -> global_load_b128 pairs,
// no divergent boundary branches: clamped address + vector select).
// Everything else (rfft2 magnitudes via LDS-staged DFT, depthwise/pointwise
// attention convs, BN) is fp32 VALU. Intermediates live in d_ws (~147MB,
// fits the 192MB global L2).
// ---------------------------------------------------------------------------

#define BB 2
#define CC 96
#define HH 128
#define WW 256
#define WF 129           // W/2+1
#define AUXC 101         // C+5
#define MIDC 24          // C/4
#define DMID 50          // AUX/2
#define HWSZ (HH*WW)     // 32768
#define WFH (HH*WF)      // 16512

typedef __attribute__((ext_vector_type(16))) __bf16 v16bf;
typedef __attribute__((ext_vector_type(8)))  float  v8f;

__device__ __forceinline__ unsigned short f2bf_bits(float f) {
  union { float f; unsigned u; } v; v.f = f;
  unsigned r = (v.u + 0x7FFFu + ((v.u >> 16) & 1u)) >> 16;   // RNE
  return (unsigned short)r;
}
__device__ __forceinline__ float bf2f(unsigned short s) {
  union { unsigned u; float f; } v; v.u = ((unsigned)s) << 16; return v.f;
}
__device__ __forceinline__ v16bf load_frag16(const unsigned short* p) {
  v16bf r;
#pragma unroll
  for (int i = 0; i < 16; ++i) {
    union { unsigned short s; __bf16 b; } u; u.s = p[i]; r[i] = u.b;
  }
  return r;
}
__device__ __forceinline__ float sigmoidf_(float x) { return 1.f / (1.f + expf(-x)); }

// ---------------- FFT helpers (row rfft 256pt, column DFT 128pt) -----------

__global__ void k_rowdft(const float* __restrict__ in, float* __restrict__ out) {
  // one block per (b,c,h); 256 threads; out complex-interleaved [bch][k][2]
  __shared__ float row[WW], tc[WW], ts[WW];
  int bch = blockIdx.x;
  int tid = threadIdx.x;
  const float* src = in + (size_t)bch * WW;
  row[tid] = src[tid];
  float sv, cv;
  __sincosf(6.28318530717958647f * (float)tid / (float)WW, &sv, &cv);
  tc[tid] = cv; ts[tid] = sv;
  __syncthreads();
  if (tid < WF) {
    float re = 0.f, im = 0.f;
    for (int n = 0; n < WW; ++n) {
      int a = (tid * n) & (WW - 1);
      re += row[n] * tc[a];
      im -= row[n] * ts[a];
    }
    float* dst = out + ((size_t)bch * WF + tid) * 2;
    dst[0] = re; dst[1] = im;
  }
}

__global__ void k_coldft_mag(const float* __restrict__ rf, float* __restrict__ mag) {
  // one block per (b,c,k); 128 threads (one per output hf); rf [bc][h][k][2]
  __shared__ float cre[HH], cim[HH], tc[HH], ts[HH];
  int bck = blockIdx.x;
  int k  = bck % WF;
  int bc = bck / WF;
  int tid = threadIdx.x;
  const float* src = rf + (((size_t)bc * HH + tid) * WF + k) * 2;
  cre[tid] = src[0]; cim[tid] = src[1];
  float sv, cv;
  __sincosf(6.28318530717958647f * (float)tid / (float)HH, &sv, &cv);
  tc[tid] = cv; ts[tid] = sv;
  __syncthreads();
  float re = 0.f, im = 0.f;
  for (int h = 0; h < HH; ++h) {
    int a = (tid * h) & (HH - 1);
    float c0 = tc[a], s0 = ts[a];
    re += cre[h] * c0 + cim[h] * s0;
    im += cim[h] * c0 - cre[h] * s0;
  }
  const float scale = 0.00552427172801990f;   // 1/sqrt(H*W), norm='ortho'
  mag[((size_t)bc * HH + tid) * WF + k] = sqrtf(re * re + im * im) * scale;
}

__global__ void k_chanmean(const float* __restrict__ in, float* __restrict__ out) {
  int idx = blockIdx.x * blockDim.x + threadIdx.x;
  if (idx >= BB * WFH) return;
  int b = idx / WFH, r = idx % WFH;
  float s = 0.f;
  for (int c = 0; c < CC; ++c) s += in[((size_t)(b * CC + c)) * WFH + r];
  out[idx] = s * (1.f / CC);
}

// ---------------- frequency-domain depthwise convs -------------------------

__global__ void k_dwh7(const float* __restrict__ in, const float* __restrict__ w,
                       const float* __restrict__ bias, float* __restrict__ out) {
  int idx = blockIdx.x * blockDim.x + threadIdx.x;
  if (idx >= BB * CC * WFH) return;
  int k = idx % WF; int h = (idx / WF) % HH; int bc = idx / WFH; int c = bc % CC;
  float acc = bias[c];
  for (int i = 0; i < 7; ++i) {
    int hh = h + i - 3;
    if (hh >= 0 && hh < HH) acc += in[((size_t)bc * HH + hh) * WF + k] * w[c * 7 + i];
  }
  out[idx] = acc;
}

__global__ void k_dww7(const float* __restrict__ in, const float* __restrict__ w,
                       const float* __restrict__ bias, float* __restrict__ out) {
  int idx = blockIdx.x * blockDim.x + threadIdx.x;
  if (idx >= BB * CC * WFH) return;
  int k = idx % WF; int h = (idx / WF) % HH; int bc = idx / WFH; int c = bc % CC;
  float acc = bias[c];
  for (int i = 0; i < 7; ++i) {
    int kk = k + i - 3;
    if (kk >= 0 && kk < WF) acc += in[((size_t)bc * HH + h) * WF + kk] * w[c * 7 + i];
  }
  out[idx] = acc;
}

// ---------------- adaptive-pool gather (g1/g2/g3) --------------------------

__global__ void k_pool(const float* __restrict__ t, float* __restrict__ g) {
  int b = blockIdx.x / CC, c = blockIdx.x % CC;
  const float* src = t + (size_t)(b * CC + c) * WFH;
  float s = 0.f, p2[4] = {0,0,0,0};
  float p4[16]; for (int i = 0; i < 16; ++i) p4[i] = 0.f;
  for (int idx = threadIdx.x; idx < WFH; idx += blockDim.x) {
    int h = idx / WF, k = idx - h * WF;
    float v = src[idx];
    s += v;
    int i2 = h >> 6;
    for (int j = 0; j < 2; ++j) {
      int ws = (j * WF) / 2, we = ((j + 1) * WF + 1) / 2;
      if (k >= ws && k < we) p2[i2 * 2 + j] += v;
    }
    int i4 = h >> 5;
    for (int j = 0; j < 4; ++j) {
      int ws = (j * WF) / 4, we = ((j + 1) * WF + 3) / 4;
      if (k >= ws && k < we) p4[i4 * 4 + j] += v;
    }
  }
  __shared__ float red[256];
  __shared__ float res[21];
  for (int q = 0; q < 21; ++q) {
    float v = (q == 0) ? s : (q < 5 ? p2[q - 1] : p4[q - 5]);
    red[threadIdx.x] = v; __syncthreads();
    for (int off = 128; off > 0; off >>= 1) {
      if (threadIdx.x < (unsigned)off) red[threadIdx.x] += red[threadIdx.x + off];
      __syncthreads();
    }
    if (threadIdx.x == 0) res[q] = red[0];
    __syncthreads();
  }
  if (threadIdx.x == 0) {
    float g1 = res[0] / (float)WFH;
    float g2 = 0.f;
    for (int i = 0; i < 2; ++i)
      for (int j = 0; j < 2; ++j) {
        int ws = (j * WF) / 2, we = ((j + 1) * WF + 1) / 2;
        g2 += res[1 + i * 2 + j] / (float)(64 * (we - ws));
      }
    g2 *= 0.25f;
    float g3 = 0.f;
    for (int i = 0; i < 4; ++i)
      for (int j = 0; j < 4; ++j) {
        int ws = (j * WF) / 4, we = ((j + 1) * WF + 3) / 4;
        g3 += res[5 + i * 4 + j] / (float)(32 * (we - ws));
      }
    g3 *= (1.f / 16.f);
    g[b * 3 * CC + c] = g1;
    g[b * 3 * CC + CC + c] = g2;
    g[b * 3 * CC + 2 * CC + c] = g3;
  }
}

__global__ void k_cp1(const float* __restrict__ g, const float* __restrict__ w,
                      const float* __restrict__ bias, float* __restrict__ out) {
  int idx = blockIdx.x * blockDim.x + threadIdx.x;
  if (idx >= BB * MIDC) return;
  int b = idx / MIDC, m = idx % MIDC;
  float acc = bias[m];
  for (int i = 0; i < 3 * CC; ++i) acc += w[m * 3 * CC + i] * g[b * 3 * CC + i];
  out[idx] = fmaxf(acc, 0.f);
}

__global__ void k_cp2(const float* __restrict__ mid, const float* __restrict__ w,
                      const float* __restrict__ bias, float* __restrict__ out) {
  int idx = blockIdx.x * blockDim.x + threadIdx.x;
  if (idx >= BB * CC) return;
  int b = idx / CC, c = idx % CC;
  float acc = bias[c];
  for (int i = 0; i < MIDC; ++i) acc += w[c * MIDC + i] * mid[b * MIDC + i];
  out[idx] = sigmoidf_(acc);
}

// ---------------- spatial attention path -----------------------------------

__device__ __forceinline__ float resize_w_sample(const float* rowWF, int w) {
  float sx = (w + 0.5f) * ((float)WF / (float)WW) - 0.5f;
  float x0f = floorf(sx);
  float f = sx - x0f;
  int i0 = (int)x0f, i1 = i0 + 1;
  i0 = min(max(i0, 0), WF - 1); i1 = min(max(i1, 0), WF - 1);
  return rowWF[i0] * (1.f - f) + rowWF[i1] * f;
}

__global__ void k_smap(const float* __restrict__ x, const float* __restrict__ fe,
                       float* __restrict__ s) {
  int idx = blockIdx.x * blockDim.x + threadIdx.x;
  if (idx >= BB * HWSZ) return;
  int b = idx / HWSZ, hw = idx % HWSZ, h = hw / WW, w = hw % WW;
  float m = 0.f, q = 0.f;
  for (int c = 0; c < CC; ++c) {
    float v = x[((size_t)(b * CC + c)) * HWSZ + hw];
    m += v; q += v * v;
  }
  m *= (1.f / CC);
  q = sqrtf(q * (1.f / CC) + 1e-6f);
  float fu = resize_w_sample(fe + (size_t)b * WFH + h * WF, w);
  s[((size_t)b * 3 + 0) * HWSZ + hw] = m;
  s[((size_t)b * 3 + 1) * HWSZ + hw] = q;
  s[((size_t)b * 3 + 2) * HWSZ + hw] = fu;
}

__global__ void k_dw3(const float* __restrict__ in, const float* __restrict__ w,
                      const float* __restrict__ bias, float* __restrict__ out,
                      int taps, int vertical) {
  int idx = blockIdx.x * blockDim.x + threadIdx.x;
  if (idx >= BB * 3 * HWSZ) return;
  int hw = idx % HWSZ; int c = (idx / HWSZ) % 3; int b = idx / (3 * HWSZ);
  int h = hw / WW, ww = hw % WW;
  int R = taps / 2;
  float acc = bias[c];
  for (int i = 0; i < taps; ++i) {
    int hh = vertical ? h + i - R : h;
    int xx = vertical ? ww : ww + i - R;
    if (hh >= 0 && hh < HH && xx >= 0 && xx < WW)
      acc += in[((size_t)(b * 3 + c)) * HWSZ + hh * WW + xx] * w[c * taps + i];
  }
  out[idx] = acc;
}

__global__ void k_sr1(const float* __restrict__ in, const float* __restrict__ w,
                      const float* __restrict__ bias, float* __restrict__ out) {
  int idx = blockIdx.x * blockDim.x + threadIdx.x;
  if (idx >= BB * 3 * HWSZ) return;
  int hw = idx % HWSZ; int c = (idx / HWSZ) % 3; int b = idx / (3 * HWSZ);
  int h = hw / WW, ww = hw % WW;
  float acc = bias[c];
  for (int ky = 0; ky < 3; ++ky)
    for (int kx = 0; kx < 3; ++kx) {
      int hh = h + ky - 1, xx = ww + kx - 1;
      if (hh >= 0 && hh < HH && xx >= 0 && xx < WW)
        acc += in[((size_t)(b * 3 + c)) * HWSZ + hh * WW + xx] * w[c * 9 + ky * 3 + kx];
    }
  out[idx] = fmaxf(acc, 0.f);
}

__global__ void k_sr2(const float* __restrict__ in, const float* __restrict__ w,
                      const float* __restrict__ bias, float* __restrict__ out) {
  int idx = blockIdx.x * blockDim.x + threadIdx.x;
  if (idx >= BB * HWSZ) return;
  int b = idx / HWSZ, hw = idx % HWSZ;
  float acc = bias[0];
  for (int c = 0; c < 3; ++c) acc += in[((size_t)(b * 3 + c)) * HWSZ + hw] * w[c];
  out[idx] = sigmoidf_(acc);
}

__global__ void k_ymul(const float* __restrict__ x, const float* __restrict__ ca,
                       const float* __restrict__ sa, float* __restrict__ y) {
  int idx = blockIdx.x * blockDim.x + threadIdx.x;
  if (idx >= BB * CC * HWSZ) return;
  int b = idx / (CC * HWSZ);
  int c = (idx / HWSZ) % CC;
  int hw = idx % HWSZ;
  y[idx] = x[idx] * ca[b * CC + c] * sa[b * HWSZ + hw];
}

__global__ void k_resizew(const float* __restrict__ in, float* __restrict__ out) {
  int idx = blockIdx.x * blockDim.x + threadIdx.x;
  if (idx >= BB * HWSZ) return;
  int b = idx / HWSZ, hw = idx % HWSZ, h = hw / WW, w = hw % WW;
  out[idx] = resize_w_sample(in + (size_t)b * WFH + h * WF, w);
}

// ---------------- NCHW f32 -> NHWC bf16 repack for WMMA --------------------

__global__ void k_nhwc(const float* __restrict__ y, unsigned short* __restrict__ o) {
  // o[b][h][w][c], thread idx has c fastest -> coalesced 2B writes
  int idx = blockIdx.x * blockDim.x + threadIdx.x;
  if (idx >= BB * HWSZ * CC) return;
  int c = idx % CC;
  int hw = (idx / CC) % HWSZ;
  int b = idx / (CC * HWSZ);
  o[idx] = f2bf_bits(y[((size_t)(b * CC + c)) * HWSZ + hw]);
}

// ---------------- offset / gate convs over aux channels --------------------

__global__ void k_offconv(const float* __restrict__ y, const float* __restrict__ fe2,
                          const float* __restrict__ wgt, const float* __restrict__ bias,
                          float* __restrict__ out) {
  int idx = blockIdx.x * blockDim.x + threadIdx.x;
  if (idx >= BB * 18 * HWSZ) return;
  int hw = idx % HWSZ; int oc = (idx / HWSZ) % 18; int b = idx / (18 * HWSZ);
  int h = hw / WW, w = hw % WW;
  float st[3], ct[3], sp[3], cp[3]; bool vx[3], vy[3];
  for (int d = 0; d < 3; ++d) {
    int ww = w + d - 1; vx[d] = (ww >= 0 && ww < WW);
    float th = -3.14159265358979324f + (float)ww * (6.28318530717958647f / (WW - 1));
    __sincosf(th, &st[d], &ct[d]);
    int hh = h + d - 1; vy[d] = (hh >= 0 && hh < HH);
    float ph = -1.57079632679489662f + (float)hh * (3.14159265358979324f / (HH - 1));
    __sincosf(ph, &sp[d], &cp[d]);
  }
  float acc = bias[oc];
  const float* wp = wgt + (size_t)oc * AUXC * 9;
  for (int ci = 0; ci < AUXC; ++ci)
    for (int ky = 0; ky < 3; ++ky) {
      if (!vy[ky]) continue;
      int hh = h + ky - 1;
      for (int kx = 0; kx < 3; ++kx) {
        if (!vx[kx]) continue;
        int ww = w + kx - 1;
        float a;
        if (ci < CC)            a = y[(((size_t)(b * CC + ci)) * HH + hh) * WW + ww];
        else if (ci == CC)      a = st[kx];
        else if (ci == CC + 1)  a = ct[kx];
        else if (ci == CC + 2)  a = sp[ky];
        else if (ci == CC + 3)  a = cp[ky];
        else                    a = fe2[(size_t)b * HWSZ + hh * WW + ww];
        acc += a * wp[(ci * 3 + ky) * 3 + kx];
      }
    }
  out[idx] = acc;
}

__global__ void k_gate(const float* __restrict__ y, const float* __restrict__ fe2,
                       const float* __restrict__ w1, const float* __restrict__ b1,
                       const float* __restrict__ w2, const float* __restrict__ b2,
                       float* __restrict__ gate) {
  int idx = blockIdx.x * blockDim.x + threadIdx.x;
  if (idx >= BB * HWSZ) return;
  int b = idx / HWSZ, hw = idx % HWSZ, h = hw / WW, w = hw % WW;
  float st[3], ct[3], sp[3], cp[3]; bool vx[3], vy[3];
  for (int d = 0; d < 3; ++d) {
    int ww = w + d - 1; vx[d] = (ww >= 0 && ww < WW);
    float th = -3.14159265358979324f + (float)ww * (6.28318530717958647f / (WW - 1));
    __sincosf(th, &st[d], &ct[d]);
    int hh = h + d - 1; vy[d] = (hh >= 0 && hh < HH);
    float ph = -1.57079632679489662f + (float)hh * (3.14159265358979324f / (HH - 1));
    __sincosf(ph, &sp[d], &cp[d]);
  }
  float accm[DMID];
#pragma unroll
  for (int m = 0; m < DMID; ++m) accm[m] = b1[m];
  for (int ci = 0; ci < AUXC; ++ci)
    for (int ky = 0; ky < 3; ++ky) {
      if (!vy[ky]) continue;
      int hh = h + ky - 1;
      for (int kx = 0; kx < 3; ++kx) {
        if (!vx[kx]) continue;
        int ww = w + kx - 1;
        float a;
        if (ci < CC)            a = y[(((size_t)(b * CC + ci)) * HH + hh) * WW + ww];
        else if (ci == CC)      a = st[kx];
        else if (ci == CC + 1)  a = ct[kx];
        else if (ci == CC + 2)  a = sp[ky];
        else if (ci == CC + 3)  a = cp[ky];
        else                    a = fe2[(size_t)b * HWSZ + hh * WW + ww];
#pragma unroll
        for (int m = 0; m < DMID; ++m)
          accm[m] += a * w1[(((size_t)m * AUXC + ci) * 3 + ky) * 3 + kx];
      }
    }
  float gsum = b2[0];
#pragma unroll
  for (int m = 0; m < DMID; ++m) gsum += fmaxf(accm[m], 0.f) * w2[m];
  gate[idx] = sigmoidf_(gsum);
}

// ---------------- WMMA weight packing + implicit-GEMM convs ----------------
// Roles: A = weights (M = co), B = activations (N = pixel).
// A-fragment layout (16x32 bf16): M = lane%16, K = 16*(lane/16)+e (same order as B).
// B-fragment layout (32x16 bf16): N = lane%16, K = 16*(lane/16)+e
//   -> 16 contiguous channels in NHWC bf16 = one 32B contiguous run per lane.
// C/D f32 16x16: N(pixel) = lane%16, M(co) = v + 8*(lane/16).

__global__ void k_pack(const float* __restrict__ wsrc, unsigned short* __restrict__ dst) {
  // wsrc flat [co][ci][9] (== [C][C][3][3] with tap = ky*3+kx); A-layout pack
  int tid = blockIdx.x * blockDim.x + threadIdx.x;
  if (tid >= 9 * 3 * 6 * 32) return;
  int lane = tid & 31;
  int wi = tid >> 5;                // = (tap*3 + kt)*6 + ntile
  int ntile = wi % 6;
  int kt = (wi / 6) % 3;
  int tap = wi / 18;
  int m = lane & 15, hl = lane >> 4;
  int co = ntile * 16 + m;
  for (int e = 0; e < 16; ++e) {
    int K = hl * 16 + e;
    int ci = kt * 32 + K;
    float v = wsrc[((size_t)co * CC + ci) * 9 + tap];
    dst[(size_t)tid * 16 + e] = f2bf_bits(v);
  }
}

__global__ __launch_bounds__(128) void k_wmma_base(const unsigned short* __restrict__ yh,
                                                   const unsigned short* __restrict__ wp,
                                                   const float* __restrict__ bias,
                                                   float* __restrict__ out) {
  int wave = threadIdx.x >> 5, lane = threadIdx.x & 31;
  int tile = blockIdx.x * 4 + wave;     // 0..8191 : (b, h, w-tile)
  int b  = tile >> 11;
  int h  = (tile >> 4) & (HH - 1);
  int w0 = (tile & 15) << 4;
  int n = lane & 15, hl = lane >> 4;    // n = pixel within tile
  v16bf zf;
#pragma unroll
  for (int i = 0; i < 16; ++i) zf[i] = (__bf16)0.f;
  v8f zero = {0.f,0.f,0.f,0.f,0.f,0.f,0.f,0.f};
  v8f acc[6] = {zero, zero, zero, zero, zero, zero};
  for (int tap = 0; tap < 9; ++tap) {
    int dy = tap / 3 - 1, dx = tap % 3 - 1;
    int hs = h + dy, ws = w0 + n + dx;
    bool inb = (hs >= 0) && (hs < HH) && (ws >= 0) && (ws < WW);
    int hc = min(max(hs, 0), HH - 1), wc = min(max(ws, 0), WW - 1);
    const unsigned short* act =
        yh + (((size_t)b * HH + hc) * WW + wc) * CC + hl * 16;
    for (int kt = 0; kt < 3; ++kt) {
      v16bf bfr = load_frag16(act + kt * 32);   // 32B contiguous per lane
      bfr = inb ? bfr : zf;                     // vector select, no branch
      const unsigned short* wq = wp + (size_t)((tap * 3 + kt) * 6) * 512;
#pragma unroll
      for (int nt = 0; nt < 6; ++nt) {
        v16bf af = load_frag16(wq + (nt * 32 + lane) * 16);
        acc[nt] = __builtin_amdgcn_wmma_f32_16x16x32_bf16(
            false, af, false, bfr, (short)0, acc[nt], false, false);
      }
    }
  }
#pragma unroll
  for (int nt = 0; nt < 6; ++nt) {
#pragma unroll
    for (int v = 0; v < 8; ++v) {
      int co = nt * 16 + hl * 8 + v;
      out[(((size_t)(b * CC + co)) * HH + h) * WW + w0 + n] = acc[nt][v] + bias[co];
    }
  }
}

__global__ __launch_bounds__(128) void k_wmma_def(const unsigned short* __restrict__ yh,
                                                  const float* __restrict__ offs,
                                                  const unsigned short* __restrict__ wp,
                                                  const float* __restrict__ bias,
                                                  float* __restrict__ out) {
  int wave = threadIdx.x >> 5, lane = threadIdx.x & 31;
  int tile = blockIdx.x * 4 + wave;
  int b  = tile >> 11;
  int h  = (tile >> 4) & (HH - 1);
  int w0 = (tile & 15) << 4;
  int n = lane & 15, hl = lane >> 4;
  int wpx = w0 + n;
  v8f zero = {0.f,0.f,0.f,0.f,0.f,0.f,0.f,0.f};
  v8f acc[6] = {zero, zero, zero, zero, zero, zero};
  const unsigned short* ybh = yh + (size_t)b * HWSZ * CC;
  for (int tap = 0; tap < 9; ++tap) {
    float ox = offs[(((size_t)(b * 18 + 2 * tap)) * HH + h) * WW + wpx];
    float oy = offs[(((size_t)(b * 18 + 2 * tap + 1)) * HH + h) * WW + wpx];
    float px = (float)wpx + ox, py = (float)h + oy;
    float fx0 = floorf(px), fy0 = floorf(py);
    float wx = px - fx0, wy = py - fy0;
    int x0 = (int)fx0, y0 = (int)fy0;
    const unsigned short* nb[4]; float wt4[4];
#pragma unroll
    for (int jy = 0; jy < 2; ++jy)
#pragma unroll
      for (int jx = 0; jx < 2; ++jx) {
        int xi = x0 + jx, yi = y0 + jy;
        bool valid = (xi >= 0) && (xi < WW) && (yi >= 0) && (yi < HH);
        int xc = min(max(xi, 0), WW - 1), yc = min(max(yi, 0), HH - 1);
        int q = jy * 2 + jx;
        nb[q] = ybh + ((size_t)yc * WW + xc) * CC + hl * 16;
        float g = (jx ? wx : 1.f - wx) * (jy ? wy : 1.f - wy);
        wt4[q] = valid ? g : 0.f;
      }
    for (int kt = 0; kt < 3; ++kt) {
      const unsigned short* p0 = nb[0] + kt * 32;
      const unsigned short* p1 = nb[1] + kt * 32;
      const unsigned short* p2 = nb[2] + kt * 32;
      const unsigned short* p3 = nb[3] + kt * 32;
      v16bf bfr;
#pragma unroll
      for (int e = 0; e < 16; ++e) {
        float v = wt4[0] * bf2f(p0[e]) + wt4[1] * bf2f(p1[e])
                + wt4[2] * bf2f(p2[e]) + wt4[3] * bf2f(p3[e]);
        bfr[e] = (__bf16)v;
      }
      const unsigned short* wq = wp + (size_t)((tap * 3 + kt) * 6) * 512;
#pragma unroll
      for (int nt = 0; nt < 6; ++nt) {
        v16bf af = load_frag16(wq + (nt * 32 + lane) * 16);
        acc[nt] = __builtin_amdgcn_wmma_f32_16x16x32_bf16(
            false, af, false, bfr, (short)0, acc[nt], false, false);
      }
    }
  }
#pragma unroll
  for (int nt = 0; nt < 6; ++nt) {
#pragma unroll
    for (int v = 0; v < 8; ++v) {
      int co = nt * 16 + hl * 8 + v;
      out[(((size_t)(b * CC + co)) * HH + h) * WW + w0 + n] = acc[nt][v] + bias[co];
    }
  }
}

// ---------------- blend + batchnorm + residual relu ------------------------

__global__ void k_blend(float* __restrict__ zb, const float* __restrict__ yd,
                        const float* __restrict__ gate) {
  int idx = blockIdx.x * blockDim.x + threadIdx.x;
  if (idx >= BB * CC * HWSZ) return;
  int b = idx / (CC * HWSZ), hw = idx % HWSZ;
  float g = gate[b * HWSZ + hw];
  zb[idx] = (1.f - g) * zb[idx] + g * yd[idx];
}

__global__ void k_bnstats(const float* __restrict__ z, float* __restrict__ stats) {
  int c = blockIdx.x;
  float s = 0.f, q = 0.f;
  for (int i = threadIdx.x; i < BB * HWSZ; i += blockDim.x) {
    int b = i / HWSZ, hw = i % HWSZ;
    float v = z[((size_t)(b * CC + c)) * HWSZ + hw];
    s += v; q += v * v;
  }
  __shared__ float rs[256], rq[256];
  rs[threadIdx.x] = s; rq[threadIdx.x] = q; __syncthreads();
  for (int o = 128; o > 0; o >>= 1) {
    if (threadIdx.x < (unsigned)o) {
      rs[threadIdx.x] += rs[threadIdx.x + o];
      rq[threadIdx.x] += rq[threadIdx.x + o];
    }
    __syncthreads();
  }
  if (threadIdx.x == 0) {
    float nfl = (float)(BB * HWSZ);
    float mu = rs[0] / nfl;
    float var = rq[0] / nfl - mu * mu;
    stats[2 * c] = mu;
    stats[2 * c + 1] = rsqrtf(var + 1e-5f);
  }
}

__global__ void k_final(const float* __restrict__ z, const float* __restrict__ x,
                        const float* __restrict__ gm, const float* __restrict__ bt,
                        const float* __restrict__ stats, float* __restrict__ out) {
  int idx = blockIdx.x * blockDim.x + threadIdx.x;
  if (idx >= BB * CC * HWSZ) return;
  int c = (idx / HWSZ) % CC;
  float zn = (z[idx] - stats[2 * c]) * stats[2 * c + 1] * gm[c] + bt[c];
  out[idx] = fmaxf(zn + x[idx], 0.f);
}

// ---------------------------------------------------------------------------

extern "C" void kernel_launch(void* const* d_in, const int* in_sizes, int n_in,
                              void* d_out, int out_size, void* d_ws, size_t ws_size,
                              hipStream_t stream) {
  const float* x     = (const float*)d_in[0];
  const float* fdh_w = (const float*)d_in[1];
  const float* fdh_b = (const float*)d_in[2];
  const float* fdw_w = (const float*)d_in[3];
  const float* fdw_b = (const float*)d_in[4];
  const float* cp1_w = (const float*)d_in[5];
  const float* cp1_b = (const float*)d_in[6];
  const float* cp2_w = (const float*)d_in[7];
  const float* cp2_b = (const float*)d_in[8];
  const float* sdh_w = (const float*)d_in[9];
  const float* sdh_b = (const float*)d_in[10];
  const float* sdw_w = (const float*)d_in[11];
  const float* sdw_b = (const float*)d_in[12];
  const float* sr1_w = (const float*)d_in[13];
  const float* sr1_b = (const float*)d_in[14];
  const float* sr2_w = (const float*)d_in[15];
  const float* sr2_b = (const float*)d_in[16];
  const float* off_w = (const float*)d_in[17];
  const float* off_b = (const float*)d_in[18];
  const float* dg1_w = (const float*)d_in[19];
  const float* dg1_b = (const float*)d_in[20];
  const float* dg2_w = (const float*)d_in[21];
  const float* dg2_b = (const float*)d_in[22];
  const float* base_w = (const float*)d_in[23];
  const float* base_b = (const float*)d_in[24];
  const float* def_w  = (const float*)d_in[25];
  const float* def_b  = (const float*)d_in[26];
  const float* bn_g   = (const float*)d_in[27];
  const float* bn_b   = (const float*)d_in[28];
  float* out = (float*)d_out;

  char* ws = (char*)d_ws;
  size_t off = 0;
  auto alloc = [&](size_t bytes) -> char* {
    char* p = ws + off;
    off = (off + bytes + 255) & ~(size_t)255;
    return p;
  };
  const size_t BCWFH = (size_t)BB * CC * WFH;
  float* rowfft = (float*)alloc(2 * BCWFH * 4);     // also reused as t1
  float* mag    = (float*)alloc(BCWFH * 4);
  float* tbuf   = (float*)alloc(BCWFH * 4);
  float* fe     = (float*)alloc((size_t)BB * WFH * 4);
  float* magy   = (float*)alloc((size_t)BB * WFH * 4);
  float* gbuf   = (float*)alloc((size_t)BB * 3 * CC * 4);
  float* camid  = (float*)alloc((size_t)BB * MIDC * 4);
  float* ca     = (float*)alloc((size_t)BB * CC * 4);
  float* sbuf0  = (float*)alloc((size_t)BB * 3 * HWSZ * 4);
  float* sbuf1  = (float*)alloc((size_t)BB * 3 * HWSZ * 4);
  float* sa     = (float*)alloc((size_t)BB * HWSZ * 4);
  float* ybuf   = (float*)alloc((size_t)BB * CC * HWSZ * 4);
  float* freqe  = (float*)alloc((size_t)BB * HWSZ * 4);
  float* offbuf = (float*)alloc((size_t)BB * 18 * HWSZ * 4);
  float* gate   = (float*)alloc((size_t)BB * HWSZ * 4);
  float* ybase  = (float*)alloc((size_t)BB * CC * HWSZ * 4);   // becomes z
  float* ydef   = (float*)alloc((size_t)BB * CC * HWSZ * 4);
  unsigned short* ynhwc  = (unsigned short*)alloc((size_t)BB * HWSZ * CC * 2);
  unsigned short* wpackA = (unsigned short*)alloc(9 * 3 * 6 * 32 * 16 * 2);
  unsigned short* wpackD = (unsigned short*)alloc(9 * 3 * 6 * 32 * 16 * 2);
  float* stats = (float*)alloc(2 * CC * 4);
  float* t1 = rowfft;   // alias: free between the two FFT passes

  const int TB = 256;
  auto grid = [&](long n) { return (int)((n + TB - 1) / TB); };

  // weight packs (independent)
  k_pack<<<grid(9 * 3 * 6 * 32), TB, 0, stream>>>(base_w, wpackA);
  k_pack<<<grid(9 * 3 * 6 * 32), TB, 0, stream>>>(def_w, wpackD);

  // ---- frequency attention ----
  k_rowdft<<<BB * CC * HH, 256, 0, stream>>>(x, rowfft);
  k_coldft_mag<<<BB * CC * WF, 128, 0, stream>>>(rowfft, mag);
  k_chanmean<<<grid((long)BB * WFH), TB, 0, stream>>>(mag, fe);
  k_dwh7<<<grid((long)BB * CC * WFH), TB, 0, stream>>>(mag, fdh_w, fdh_b, t1);
  k_dww7<<<grid((long)BB * CC * WFH), TB, 0, stream>>>(t1, fdw_w, fdw_b, tbuf);
  k_pool<<<BB * CC, 256, 0, stream>>>(tbuf, gbuf);
  k_cp1<<<grid(BB * MIDC), TB, 0, stream>>>(gbuf, cp1_w, cp1_b, camid);
  k_cp2<<<grid(BB * CC), TB, 0, stream>>>(camid, cp2_w, cp2_b, ca);

  // ---- spatial attention ----
  k_smap<<<grid((long)BB * HWSZ), TB, 0, stream>>>(x, fe, sbuf0);
  k_dw3<<<grid((long)BB * 3 * HWSZ), TB, 0, stream>>>(sbuf0, sdh_w, sdh_b, sbuf1, 15, 1);
  k_dw3<<<grid((long)BB * 3 * HWSZ), TB, 0, stream>>>(sbuf1, sdw_w, sdw_b, sbuf0, 15, 0);
  k_sr1<<<grid((long)BB * 3 * HWSZ), TB, 0, stream>>>(sbuf0, sr1_w, sr1_b, sbuf1);
  k_sr2<<<grid((long)BB * HWSZ), TB, 0, stream>>>(sbuf1, sr2_w, sr2_b, sa);
  k_ymul<<<grid((long)BB * CC * HWSZ), TB, 0, stream>>>(x, ca, sa, ybuf);

  // ---- rfft2(y) magnitude -> freq_e ----
  k_rowdft<<<BB * CC * HH, 256, 0, stream>>>(ybuf, rowfft);
  k_coldft_mag<<<BB * CC * WF, 128, 0, stream>>>(rowfft, mag);
  k_chanmean<<<grid((long)BB * WFH), TB, 0, stream>>>(mag, magy);
  k_resizew<<<grid((long)BB * HWSZ), TB, 0, stream>>>(magy, freqe);

  // ---- offsets + gate + NHWC bf16 repack ----
  k_offconv<<<grid((long)BB * 18 * HWSZ), TB, 0, stream>>>(ybuf, freqe, off_w, off_b, offbuf);
  k_gate<<<grid((long)BB * HWSZ), TB, 0, stream>>>(ybuf, freqe, dg1_w, dg1_b, dg2_w, dg2_b, gate);
  k_nhwc<<<grid((long)BB * HWSZ * CC), TB, 0, stream>>>(ybuf, ynhwc);

  // ---- heavy convs on WMMA bf16 ----
  k_wmma_base<<<2048, 128, 0, stream>>>(ynhwc, wpackA, base_b, ybase);
  k_wmma_def<<<2048, 128, 0, stream>>>(ynhwc, offbuf, wpackD, def_b, ydef);

  // ---- blend + BN + residual relu ----
  k_blend<<<grid((long)BB * CC * HWSZ), TB, 0, stream>>>(ybase, ydef, gate);
  k_bnstats<<<CC, 256, 0, stream>>>(ybase, stats);
  k_final<<<grid((long)BB * CC * HWSZ), TB, 0, stream>>>(ybase, x, bn_g, bn_b, stats, out);
  (void)in_sizes; (void)n_in; (void)out_size; (void)ws_size;
}
